// MultiDirectionalSpatialScanner_82824149336673
// MI455X (gfx1250) — matrix operand order repaired and checked by reading, not compiled, not run on hardware
//
#include <hip/hip_runtime.h>
#include <math.h>

typedef __bf16 bf16;
typedef __attribute__((ext_vector_type(8)))  __bf16 v8bf;
typedef __attribute__((ext_vector_type(16))) __bf16 v16bf;
typedef __attribute__((ext_vector_type(8)))  float  v8f;

#define NTOK   32768   // B*H*W = 32*32*32
#define DMODEL 384
#define LSEQ   1024
#define NBATCH 32
#define DSTATE 4
#define DTRANK 24

// GEMM tiling: 128x128 block tile, 8 wave32 waves, each 32x64 (2x4 WMMA 16x16), K-step 32
#define BM 128
#define BN 128
#define BK 32

enum { ACT_NONE = 0, ACT_SILU = 1, ACT_SOFTPLUS = 2 };
enum { MAP_NONE = 0, MAP_FLIPW = 1 };

__device__ __forceinline__ int map_row(int r, int mode) {
  if (mode == MAP_FLIPW) {
    int l = r & 1023, b = r >> 10;
    int hh = l >> 5, w = l & 31;
    return (b << 10) + (hh << 5) + (31 - w);
  }
  return r;
}

__device__ __forceinline__ float sigmoidf_fast(float v) { return 1.f / (1.f + __expf(-v)); }

// global->LDS tile loaders (16 halves), overloaded on source type
__device__ __forceinline__ void load16(const bf16* __restrict__ src, bf16* dst) {
  *(v8bf*)(dst)     = *(const v8bf*)(src);
  *(v8bf*)(dst + 8) = *(const v8bf*)(src + 8);
}
__device__ __forceinline__ void load16(const float* __restrict__ src, bf16* dst) {
  #pragma unroll
  for (int j = 0; j < 16; ++j) dst[j] = (bf16)src[j];
}

// ---------------------------------------------------------------------------
// WMMA GEMM: C[M,N] = act(A[M,K] @ W[K,N] (+ bias)), bf16 math, f32 accum.
// Weights are pre-transposed: BwT is [N][K] row-major (k contiguous), ldbt = K.
// AMAP flips width of A's token-row index; OMAP flips width of the output row.
// KAL: K % BK == 0 (vector loads, no guards). NAL: N % BN == 0 (no col guards).
// M must be a multiple of BM.
// ---------------------------------------------------------------------------
template<typename AT, typename OT, int ACT, int AMAP, int OMAP, bool KAL, bool NAL>
__global__ __launch_bounds__(256) void gemm_wmma(
    const AT* __restrict__ A, int lda,
    const bf16* __restrict__ BwT, int ldbt,
    OT* __restrict__ C, int ldc, int c_col0,
    const float* __restrict__ bias,
    int M, int N, int K)
{
  __shared__ __align__(128) bf16 sA[BM][BK];  // 8 KB
  __shared__ __align__(128) bf16 sB[BN][BK];  // 8 KB, [n][k]

  const int tid  = threadIdx.x;
  const int wave = tid >> 5;
  const int lane = tid & 31;
  const int bm = blockIdx.y * BM;
  const int bn = blockIdx.x * BN;
  const int wm = (wave & 3) * 32;
  const int wn = (wave >> 2) * 64;
  const int g  = lane >> 4;     // lane half-group (ISA 16-bit A/B layout)
  const int lm = lane & 15;

  v8f acc[2][4];
  #pragma unroll
  for (int r = 0; r < 2; ++r)
    #pragma unroll
    for (int c = 0; c < 4; ++c)
      #pragma unroll
      for (int e = 0; e < 8; ++e) acc[r][c][e] = 0.f;

  // A-tile assignment: 256 threads -> 128 rows x 2 half-row chunks of 16
  const int a_row = tid >> 1;
  const int a_kc  = (tid & 1) << 4;
  const int a_gr  = map_row(bm + a_row, AMAP);
  const AT* a_src0 = A + (size_t)a_gr * lda + a_kc;
  // B-tile assignment: same shape (weights pre-transposed to [N][K])
  const int b_row = tid >> 1;
  const int b_kc  = (tid & 1) << 4;
  const bf16* b_src0 = BwT + (size_t)(bn + b_row) * ldbt + b_kc;

  for (int k0 = 0; k0 < K; k0 += BK) {
    // ---- A tile: BM x BK (row-major in LDS) ----
    {
      bf16* dst = &sA[a_row][a_kc];
      if (KAL) {
        const AT* src = a_src0 + k0;
        load16(src, dst);
        if (k0 + BK < K) __builtin_prefetch((const void*)(src + BK), 0, 3);
      } else {
        #pragma unroll
        for (int j = 0; j < 16; ++j) {
          int kk = k0 + a_kc + j;
          dst[j] = (bf16)(kk < K ? (float)a_src0[k0 + j] : 0.f);
        }
      }
    }
    // ---- B tile: BN x BK ([n][k] in LDS; k contiguous in global) ----
    {
      bf16* dst = &sB[b_row][b_kc];
      if (KAL && NAL) {
        const bf16* src = b_src0 + k0;
        load16(src, dst);
        if (k0 + BK < K) __builtin_prefetch((const void*)(src + BK), 0, 3);
      } else {
        const int n = bn + b_row;
        #pragma unroll
        for (int j = 0; j < 16; ++j) {
          int kk = k0 + b_kc + j;
          float v = (n < N && kk < K) ? (float)BwT[(size_t)n * ldbt + kk] : 0.f;
          dst[j] = (bf16)v;
        }
      }
    }
    __syncthreads();

    // Fragments per ISA 16-bit layouts:
    // A 16x32: lane (g,lm): M=lm, elems 0..7 = K[g*8..g*8+7], elems 8..15 = K[16+g*8 ..]
    // B 32x16 ([n][k] LDS): lane (g,lm): N=lm, elems 0..7 = K[g*8..g*8+7], elems 8..15 = K[16+g*8 ..]
    v16bf af[2], bfr[4];
    #pragma unroll
    for (int r = 0; r < 2; ++r) {
      const bf16* pa = &sA[wm + r * 16 + lm][0];
      v8bf lo = *(const v8bf*)(pa + (g << 3));
      v8bf hi = *(const v8bf*)(pa + 16 + (g << 3));
      af[r] = __builtin_shufflevector(lo, hi, 0,1,2,3,4,5,6,7,8,9,10,11,12,13,14,15);
    }
    #pragma unroll
    for (int c = 0; c < 4; ++c) {
      const bf16* pb = &sB[wn + c * 16 + lm][0];
      v8bf blo = *(const v8bf*)(pb + (g << 3));
      v8bf bhi = *(const v8bf*)(pb + 16 + (g << 3));
      bfr[c] = __builtin_shufflevector(blo, bhi, 0,1,2,3,4,5,6,7,8,9,10,11,12,13,14,15);
    }
    #pragma unroll
    for (int r = 0; r < 2; ++r)
      #pragma unroll
      for (int c = 0; c < 4; ++c)
        acc[r][c] = __builtin_amdgcn_wmma_f32_16x16x32_bf16(
            false, af[r], false, bfr[c], (short)0, acc[r][c], false, false);
    __syncthreads();
  }

  // ---- store: C/D 16x16 f32 layout: VGPR e -> M = e + g*8, N = lm ----
  #pragma unroll
  for (int r = 0; r < 2; ++r) {
    const int row  = bm + wm + r * 16 + (g << 3);
    #pragma unroll
    for (int c = 0; c < 4; ++c) {
      const int col = bn + wn + c * 16 + lm;
      if (NAL || col < N) {
        #pragma unroll
        for (int e = 0; e < 8; ++e) {
          float v = acc[r][c][e];
          if (bias) v += bias[col];
          if (ACT == ACT_SILU)          v = v * sigmoidf_fast(v);
          else if (ACT == ACT_SOFTPLUS) v = (v > 20.f) ? v : log1pf(__expf(v));
          const int orow = map_row(row + e, OMAP);
          C[(size_t)orow * ldc + c_col0 + col] = (OT)v;
        }
      }
    }
  }
}

// ---------------------------------------------------------------------------
// LayerNorm(x) + sinusoidal 2D positional embedding -> bf16
// one block per token, 128 threads x 3 elems
// ---------------------------------------------------------------------------
__global__ __launch_bounds__(128) void ln_pos_kernel(
    const float* __restrict__ x, const float* __restrict__ gg,
    const float* __restrict__ bb, bf16* __restrict__ h)
{
  const int t = blockIdx.x;
  const float* xr = x + (size_t)t * DMODEL;
  float v[3], s = 0.f, ss = 0.f;
  #pragma unroll
  for (int j = 0; j < 3; ++j) {
    v[j] = xr[threadIdx.x + j * 128];
    s += v[j]; ss += v[j] * v[j];
  }
  #pragma unroll
  for (int m = 16; m > 0; m >>= 1) {
    s  += __shfl_xor(s,  m, 32);
    ss += __shfl_xor(ss, m, 32);
  }
  __shared__ float rs[4], rss[4];
  const int wave = threadIdx.x >> 5;
  if ((threadIdx.x & 31) == 0) { rs[wave] = s; rss[wave] = ss; }
  __syncthreads();
  s  = rs[0] + rs[1] + rs[2] + rs[3];
  ss = rss[0] + rss[1] + rss[2] + rss[3];
  const float mean = s / DMODEL;
  const float var  = ss / DMODEL - mean * mean;
  const float rstd = rsqrtf(var + 1e-5f);
  const int l = t & 1023;
  const float ph = (float)(l >> 5) * (2.f / 31.f) - 1.f;
  const float pw = (float)(l & 31) * (2.f / 31.f) - 1.f;
  #pragma unroll
  for (int j = 0; j < 3; ++j) {
    const int d = threadIdx.x + j * 128;
    float y = (v[j] - mean) * rstd * gg[d] + bb[d];
    const int q = d >> 2, grp = d & 3;
    const float freq = __expf((float)(4 * q) * (-9.210340371976184f / 384.f));
    const float arg = (grp < 2 ? ph : pw) * freq;
    const float pe  = ((grp & 1) == 0) ? __sinf(arg) : __cosf(arg);
    h[(size_t)t * DMODEL + d] = (bf16)(y + pe);
  }
}

// depthwise causal conv1d (k=3) + SiLU; reads xi = xz[:, :384]
__global__ __launch_bounds__(256) void conv_silu_kernel(
    const float* __restrict__ xz, const float* __restrict__ cw,
    float* __restrict__ xcf, bf16* __restrict__ xcb)
{
  const int idx = blockIdx.x * 256 + threadIdx.x;
  if (idx >= NTOK * DMODEL) return;
  const int c = idx % DMODEL;
  const int tok = idx / DMODEL;
  const int l = tok & 1023;
  const float w0 = cw[c * 3 + 0], w1 = cw[c * 3 + 1], w2 = cw[c * 3 + 2];
  float a = w2 * xz[(size_t)tok * 768 + c];
  if (l >= 1) a += w1 * xz[(size_t)(tok - 1) * 768 + c];
  if (l >= 2) a += w0 * xz[(size_t)(tok - 2) * 768 + c];
  a = a * sigmoidf_fast(a);
  xcf[idx] = a;
  xcb[idx] = (bf16)a;
}

// selective-state scan: one thread per (batch, channel), L=1024 sequential
__global__ __launch_bounds__(DMODEL) void scan_kernel(
    const float* __restrict__ dt, const float* __restrict__ xc,
    const float* __restrict__ xdbl, const float* __restrict__ xz,
    const float* __restrict__ A_log, const float* __restrict__ Dp,
    bf16* __restrict__ yb)
{
  const int b = blockIdx.x;
  const int c = threadIdx.x;
  float Ar[DSTATE];
  #pragma unroll
  for (int s = 0; s < DSTATE; ++s) Ar[s] = -__expf(A_log[c * DSTATE + s]);
  const float Dv = Dp[c];
  float h[DSTATE] = {0.f, 0.f, 0.f, 0.f};
  const size_t base = (size_t)b * LSEQ;
  for (int l = 0; l < LSEQ; ++l) {
    const size_t tok = base + l;
    const float d  = dt[tok * DMODEL + c];
    const float xv = xc[tok * DMODEL + c];
    const float xd = xv * d;
    const float* xrow = xdbl + tok * 32;
    float y = 0.f;
    #pragma unroll
    for (int s = 0; s < DSTATE; ++s) {
      h[s] = h[s] * __expf(Ar[s] * d) + xrow[DTRANK + s] * xd;
      y += xrow[DTRANK + DSTATE + s] * h[s];
    }
    y += Dv * xv;
    const float z = xz[tok * 768 + DMODEL + c];
    y *= z * sigmoidf_fast(z);
    yb[tok * DMODEL + c] = (bf16)y;
  }
}

// WcT[d][c] = dir_weight * sum_k out_proj[c][k] * dir_proj[k][d]   (transposed store)
__global__ __launch_bounds__(256) void combine_w_kernel(
    const float* __restrict__ opw, const float* __restrict__ dirw,
    const float* __restrict__ dw, bf16* __restrict__ wcT)
{
  const int idx = blockIdx.x * 256 + threadIdx.x;
  if (idx >= DMODEL * DMODEL) return;
  const int d = idx / DMODEL, c = idx % DMODEL;
  float acc = 0.f;
  for (int k = 0; k < DMODEL; ++k) acc += opw[c * DMODEL + k] * dirw[k * DMODEL + d];
  wcT[idx] = (bf16)(acc * dw[0]);
}

// fp32 [K][N] weight -> bf16 [N][K] (transposed) one-time conversion
__global__ __launch_bounds__(256) void f2bT_kernel(const float* __restrict__ in,
                                                   bf16* __restrict__ outT,
                                                   int K, int N)
{
  const int i = blockIdx.x * 256 + threadIdx.x;
  if (i >= K * N) return;
  const int k = i / N, n = i % N;
  outT[(size_t)n * K + k] = (bf16)in[i];
}

// final: LayerNorm(residual + fused2) -> f32 output
__global__ __launch_bounds__(128) void ln_out_kernel(
    const float* __restrict__ x, const float* __restrict__ f,
    const float* __restrict__ gg, const float* __restrict__ bb,
    float* __restrict__ out)
{
  const int t = blockIdx.x;
  float v[3], s = 0.f, ss = 0.f;
  #pragma unroll
  for (int j = 0; j < 3; ++j) {
    const int d = threadIdx.x + j * 128;
    v[j] = x[(size_t)t * DMODEL + d] + f[(size_t)t * DMODEL + d];
    s += v[j]; ss += v[j] * v[j];
  }
  #pragma unroll
  for (int m = 16; m > 0; m >>= 1) {
    s  += __shfl_xor(s,  m, 32);
    ss += __shfl_xor(ss, m, 32);
  }
  __shared__ float rs[4], rss[4];
  const int wave = threadIdx.x >> 5;
  if ((threadIdx.x & 31) == 0) { rs[wave] = s; rss[wave] = ss; }
  __syncthreads();
  s  = rs[0] + rs[1] + rs[2] + rs[3];
  ss = rss[0] + rss[1] + rss[2] + rss[3];
  const float mean = s / DMODEL;
  const float rstd = rsqrtf(ss / DMODEL - mean * mean + 1e-5f);
  #pragma unroll
  for (int j = 0; j < 3; ++j) {
    const int d = threadIdx.x + j * 128;
    out[(size_t)t * DMODEL + d] = (v[j] - mean) * rstd * gg[d] + bb[d];
  }
}

// ---------------------------------------------------------------------------
extern "C" void kernel_launch(void* const* d_in, const int* in_sizes, int n_in,
                              void* d_out, int out_size, void* d_ws, size_t ws_size,
                              hipStream_t stream)
{
  const float* x        = (const float*)d_in[0];
  const float* ln_in_g  = (const float*)d_in[1];
  const float* ln_in_b  = (const float*)d_in[2];
  const float* in_projw = (const float*)d_in[3];
  const float* conv_w   = (const float*)d_in[4];
  const float* x_projw  = (const float*)d_in[5];
  const float* dt_projw = (const float*)d_in[6];
  const float* dt_projb = (const float*)d_in[7];
  const float* A_log    = (const float*)d_in[8];
  const float* D_param  = (const float*)d_in[9];
  const float* out_projw= (const float*)d_in[10];
  const float* dir_projw= (const float*)d_in[11];
  const float* fusion1  = (const float*)d_in[12];
  const float* fusion2  = (const float*)d_in[13];
  const float* ln_out_g = (const float*)d_in[14];
  const float* ln_out_b = (const float*)d_in[15];
  const float* dir_wts  = (const float*)d_in[16];
  float* out = (float*)d_out;

  // bump allocator over d_ws
  char* ws = (char*)d_ws;
  size_t off = 0;
  auto alloc = [&](size_t bytes) -> void* {
    void* p = ws + off;
    off = (off + bytes + 255) & ~(size_t)255;
    return p;
  };
  bf16*  h_b    = (bf16*) alloc((size_t)NTOK * DMODEL * 2);
  float* xz     = (float*)alloc((size_t)NTOK * 768 * 4);
  float* xc_f   = (float*)alloc((size_t)NTOK * DMODEL * 4);
  bf16*  xc_b   = (bf16*) alloc((size_t)NTOK * DMODEL * 2);
  float* xdbl   = (float*)alloc((size_t)NTOK * 32 * 4);
  float* dt     = (float*)alloc((size_t)NTOK * DMODEL * 4);
  bf16*  y_b    = (bf16*) alloc((size_t)NTOK * DMODEL * 2);
  bf16*  cat    = (bf16*) alloc((size_t)NTOK * 768 * 2);
  bf16*  fused1 = (bf16*) alloc((size_t)NTOK * 768 * 2);
  float* fused2 = (float*)alloc((size_t)NTOK * DMODEL * 4);
  // weights, bf16, pre-transposed to [N][K]
  bf16*  w_in   = (bf16*) alloc((size_t)2 * 768 * DMODEL * 2);
  bf16*  w_xp   = (bf16*) alloc((size_t)2 * 32 * DMODEL * 2);
  bf16*  w_dt   = (bf16*) alloc((size_t)2 * DMODEL * DTRANK * 2);
  bf16*  w_c    = (bf16*) alloc((size_t)2 * DMODEL * DMODEL * 2);
  bf16*  w_f1   = (bf16*) alloc((size_t)768 * 768 * 2);
  bf16*  w_f2   = (bf16*) alloc((size_t)DMODEL * 768 * 2);
  (void)ws_size; (void)in_sizes; (void)n_in; (void)out_size;

  // --- weight conversions + transposes (once per launch) ---
  auto cvtT = [&](const float* src, bf16* dstT, int K, int N) {
    f2bT_kernel<<<(K * N + 255) / 256, 256, 0, stream>>>(src, dstT, K, N);
  };
  for (int i = 0; i < 2; ++i) {
    cvtT(in_projw + (size_t)i * DMODEL * 768, w_in + (size_t)i * 768 * DMODEL, DMODEL, 768);
    cvtT(x_projw  + (size_t)i * DMODEL * 32,  w_xp + (size_t)i * 32 * DMODEL,  DMODEL, 32);
    cvtT(dt_projw + (size_t)i * DTRANK * DMODEL, w_dt + (size_t)i * DMODEL * DTRANK, DTRANK, DMODEL);
    combine_w_kernel<<<(DMODEL * DMODEL + 255) / 256, 256, 0, stream>>>(
        out_projw + (size_t)i * DMODEL * DMODEL,
        dir_projw + (size_t)i * DMODEL * DMODEL,
        dir_wts + i,
        w_c + (size_t)i * DMODEL * DMODEL);
  }
  cvtT(fusion1, w_f1, 768, 768);
  cvtT(fusion2, w_f2, 768, DMODEL);

  // --- LN + positional embedding ---
  ln_pos_kernel<<<NTOK, 128, 0, stream>>>(x, ln_in_g, ln_in_b, h_b);

  const dim3 blk(256);
  for (int i = 0; i < 2; ++i) {
    const bf16* wi = w_in + (size_t)i * 768 * DMODEL;     // [768][384]
    const bf16* wx = w_xp + (size_t)i * 32 * DMODEL;      // [32][384]
    const bf16* wd = w_dt + (size_t)i * DMODEL * DTRANK;  // [384][24]
    const bf16* wc = w_c  + (size_t)i * DMODEL * DMODEL;  // [384][384]

    // 1) xz = h @ in_proj  (direction 1 reads width-flipped rows)
    {
      dim3 grid(768 / BN, NTOK / BM);
      if (i == 0)
        gemm_wmma<bf16, float, ACT_NONE, MAP_NONE, MAP_NONE, true, true><<<grid, blk, 0, stream>>>(
            h_b, DMODEL, wi, DMODEL, xz, 768, 0, nullptr, NTOK, 768, DMODEL);
      else
        gemm_wmma<bf16, float, ACT_NONE, MAP_FLIPW, MAP_NONE, true, true><<<grid, blk, 0, stream>>>(
            h_b, DMODEL, wi, DMODEL, xz, 768, 0, nullptr, NTOK, 768, DMODEL);
    }
    // 2) depthwise causal conv + SiLU
    conv_silu_kernel<<<(NTOK * DMODEL + 255) / 256, 256, 0, stream>>>(
        xz, conv_w + (size_t)i * DMODEL * 3, xc_f, xc_b);
    // 3) x_dbl = xc @ x_proj  (N=32: col-guarded)
    {
      dim3 grid(1, NTOK / BM);
      gemm_wmma<bf16, float, ACT_NONE, MAP_NONE, MAP_NONE, true, false><<<grid, blk, 0, stream>>>(
          xc_b, DMODEL, wx, DMODEL, xdbl, 32, 0, nullptr, NTOK, 32, DMODEL);
    }
    // 4) dt = softplus(x_dbl[:, :24] @ dt_proj + bias)  (K=24: K-guarded)
    {
      dim3 grid(DMODEL / BN, NTOK / BM);
      gemm_wmma<float, float, ACT_SOFTPLUS, MAP_NONE, MAP_NONE, false, true><<<grid, blk, 0, stream>>>(
          xdbl, 32, wd, DTRANK, dt, DMODEL, 0, dt_projb + (size_t)i * DMODEL, NTOK, DMODEL, DTRANK);
    }
    // 5) selective scan + gate -> y (bf16)
    scan_kernel<<<NBATCH, DMODEL, 0, stream>>>(
        dt, xc_f, xdbl, xz,
        A_log + (size_t)i * DMODEL * DSTATE, D_param + (size_t)i * DMODEL, y_b);
    // 6) cat[:, i*384:(i+1)*384] = y @ (out_proj@dir_proj*w)  (direction 1 un-flips rows)
    {
      dim3 grid(DMODEL / BN, NTOK / BM);
      if (i == 0)
        gemm_wmma<bf16, bf16, ACT_NONE, MAP_NONE, MAP_NONE, true, true><<<grid, blk, 0, stream>>>(
            y_b, DMODEL, wc, DMODEL, cat, 768, 0, nullptr, NTOK, DMODEL, DMODEL);
      else
        gemm_wmma<bf16, bf16, ACT_NONE, MAP_NONE, MAP_FLIPW, true, true><<<grid, blk, 0, stream>>>(
            y_b, DMODEL, wc, DMODEL, cat, 768, DMODEL, nullptr, NTOK, DMODEL, DMODEL);
    }
  }

  // fusion MLP
  {
    dim3 grid(768 / BN, NTOK / BM);
    gemm_wmma<bf16, bf16, ACT_SILU, MAP_NONE, MAP_NONE, true, true><<<grid, blk, 0, stream>>>(
        cat, 768, w_f1, 768, fused1, 768, 0, nullptr, NTOK, 768, 768);
  }
  {
    dim3 grid(DMODEL / BN, NTOK / BM);
    gemm_wmma<bf16, float, ACT_NONE, MAP_NONE, MAP_NONE, true, true><<<grid, blk, 0, stream>>>(
        fused1, 768, w_f2, 768, fused2, DMODEL, 0, nullptr, NTOK, DMODEL, 768);
  }

  // final residual LayerNorm -> f32 output
  ln_out_kernel<<<NTOK, 128, 0, stream>>>(x, fused2, ln_out_g, ln_out_b, out);
}
